// WindowAttention_75806172774615
// MI455X (gfx1250) — matrix operand order repaired and checked by reading, not compiled, not run on hardware
//
#include <hip/hip_runtime.h>

typedef __attribute__((ext_vector_type(16))) _Float16 v16h;
typedef __attribute__((ext_vector_type(8)))  _Float16 v8h;
typedef __attribute__((ext_vector_type(4)))  _Float16 v4h;
typedef __attribute__((ext_vector_type(8)))  float    v8f;

#define WIN_N 49
#define NP    64
#define CDIM  128
#define C3    384
#define HEADS 4
#define DHEAD 32

// ---- LDS layout (bytes) ---------------------------------------------------
#define XS_OFF 0                               // 64x128 f16      = 16384
#define QS_OFF (NP * CDIM * 2)                 // 64x384 f16      = 49152
#define SS_OFF (QS_OFF + NP * C3 * 2)          // 4x64x64 f16     = 32768
#define VT_OFF (SS_OFF + HEADS * NP * NP * 2)  // 128x64 f16 (V^T)= 16384
#define MK_OFF (VT_OFF + CDIM * NP * 2)        // 49x49 f32       =  9604
#define LDS_BYTES (MK_OFF + ((WIN_N * WIN_N * 4 + 31) & ~31))  // 124320

// ---- fragment helpers (CDNA5 wave32 WMMA 16x16x32 f16 layouts) ------------
__device__ __forceinline__ v16h concat8(v8h lo, v8h hi) {
  return __builtin_shufflevector(lo, hi, 0, 1, 2, 3, 4, 5, 6, 7, 8, 9, 10, 11,
                                 12, 13, 14, 15);
}

// A (16x32 f16): lane<16 -> M=lane, halves0-15:K=0..15;
// lane>=16 -> K=16..31. Two aligned 16B vector loads.
__device__ __forceinline__ v16h load_a_f16(const _Float16* base, int row0,
                                           int col0, int ld, int lane) {
  const int m  = row0 + (lane & 15);
  const int k0 = col0 + ((lane >> 4) << 4);
  const _Float16* p = base + m * ld + k0;
  return concat8(*(const v8h*)p, *(const v8h*)(p + 8));
}

// B fragment from pre-packed weights: one contiguous 32B load per lane.
__device__ __forceinline__ v16h load_b_packed(const _Float16* pk, int kt,
                                              int nt, int ktn, int lane) {
  return *(const v16h*)(pk + (size_t)((nt * ktn + kt) * 32 + lane) * 16);
}

__device__ __forceinline__ v8f wmma_f16(v16h a, v16h b, v8f c) {
  return __builtin_amdgcn_wmma_f32_16x16x32_f16(false, a, false, b, (short)0,
                                                c, false, false);
}

// ---- prep: convert + pack weights into WMMA B-fragment order --------------
// packed[(nt*KT + kt)*32 + lane][i] = W[kt*32 + (lane>=16)*16 + i][nt*16 + (lane&15)]
__global__ void pack_weights_kernel(const float* __restrict__ wq,
                                    const float* __restrict__ wp,
                                    _Float16* __restrict__ dst) {
  const int p = blockIdx.x * 256 + threadIdx.x;
  if (p < CDIM * C3) {  // Wqkv: KT=4, NT=24
    const int i = p & 15, lane = (p >> 4) & 31, f = p >> 9;
    const int kt = f & 3, nt = f >> 2;
    const int row = kt * 32 + ((lane >> 4) << 4) + i;
    const int col = nt * 16 + (lane & 15);
    dst[p] = (_Float16)wq[row * C3 + col];
  } else {  // Wproj: KT=4, NT=8
    const int q = p - CDIM * C3;
    if (q < CDIM * CDIM) {
      const int i = q & 15, lane = (q >> 4) & 31, f = q >> 9;
      const int kt = f & 3, nt = f >> 2;
      const int row = kt * 32 + ((lane >> 4) << 4) + i;
      const int col = nt * 16 + (lane & 15);
      dst[CDIM * C3 + q] = (_Float16)wp[row * CDIM + col];
    }
  }
}

// ---- fused window-attention kernel: one block (8 waves) per window --------
__global__ __launch_bounds__(256) void win_attn_kernel(
    const float* __restrict__ x, const float* __restrict__ mask,
    const float* __restrict__ bqkv, const float* __restrict__ bproj,
    const _Float16* __restrict__ wq16, const _Float16* __restrict__ wp16,
    float* __restrict__ out) {
  extern __shared__ char smem[];
  _Float16* xs = (_Float16*)(smem + XS_OFF);  // x tile f16, rows padded to 64
  _Float16* qs = (_Float16*)(smem + QS_OFF);  // qkv (O overwrites V region)
  _Float16* ss = (_Float16*)(smem + SS_OFF);  // scores / P, per head 64x64
  _Float16* vt = (_Float16*)(smem + VT_OFF);  // V^T: [d=128][token=64]
  float*    mk = (float*)(smem + MK_OFF);     // window mask 49x49 f32

  const int b    = blockIdx.x;
  const int tid  = threadIdx.x;
  const int wave = tid >> 5;   // 0..7
  const int lane = tid & 31;

  // ---- Stage 1: x -> LDS f16 (pad rows), stage mask -> LDS ----
  {
    const float* xb = x + (size_t)b * WIN_N * CDIM;
#pragma unroll
    for (int i = 0; i < 8; ++i) {
      const int slot = tid + i * 256;  // 2048 float4 slots
      const int r = slot >> 5;
      const int c = (slot & 31) << 2;
      float4 v = make_float4(0.f, 0.f, 0.f, 0.f);
      if (r < WIN_N) v = *(const float4*)(xb + r * CDIM + c);
      v4h hv;
      hv[0] = (_Float16)v.x; hv[1] = (_Float16)v.y;
      hv[2] = (_Float16)v.z; hv[3] = (_Float16)v.w;
      *(v4h*)(xs + r * CDIM + c) = hv;
    }
    const float* mw = mask + (size_t)(b & 63) * WIN_N * WIN_N;
    for (int idx = tid; idx < WIN_N * WIN_N; idx += 256) mk[idx] = mw[idx];
  }
  __syncthreads();

  // ---- Stage 2: qkv = xs @ Wqkv + bqkv ----
  // wave owns m-tile (wave&3) and half the n-range (wave>>2); A hoisted, 12x reuse
  {
    const int mt = wave & 3;
    const int g  = wave >> 2;
    const int m0 = mt << 4;
    v16h a[4];
#pragma unroll
    for (int kt = 0; kt < 4; ++kt)
      a[kt] = load_a_f16(xs, m0, kt << 5, CDIM, lane);
    const int mbase = m0 + ((lane >> 4) << 3);

    // Q,K tiles: nt = g*8 .. g*8+7 -> qs (strided scalar stores)
#pragma unroll
    for (int j = 0; j < 8; ++j) {
      const int nt = (g << 3) + j;
      v8f acc = {};
#pragma unroll
      for (int kt = 0; kt < 4; ++kt)
        acc = wmma_f16(a[kt], load_b_packed(wq16, kt, nt, 4, lane), acc);
      const int   n    = (nt << 4) + (lane & 15);
      const float bias = bqkv[n];
#pragma unroll
      for (int r = 0; r < 8; ++r)
        qs[(mbase + r) * C3 + n] = (_Float16)(acc[r] + bias);
    }
    // V tiles: nt = 16 + g*4 .. +3 -> vt transposed, one 16B store per lane
#pragma unroll
    for (int j = 0; j < 4; ++j) {
      const int nt = 16 + (g << 2) + j;
      v8f acc = {};
#pragma unroll
      for (int kt = 0; kt < 4; ++kt)
        acc = wmma_f16(a[kt], load_b_packed(wq16, kt, nt, 4, lane), acc);
      const int   n    = (nt << 4) + (lane & 15);
      const float bias = bqkv[n];
      const int   d    = n - 2 * CDIM;
      v8h vv;
#pragma unroll
      for (int r = 0; r < 8; ++r) vv[r] = (_Float16)(acc[r] + bias);
      *(v8h*)(vt + d * NP + mbase) = vv;
    }
  }
  __syncthreads();

  // ---- Stages 3-4: attention. 2 waves per head; sub owns m-tiles/rows so
  //      score->softmax->PV stays wave-local (no extra barriers needed). ----
  const int   h     = wave & 3;   // head
  const int   sub   = wave >> 2;  // 0: rows 0-31 (mt 0,1); 1: rows 32-63 (mt 2,3)
  _Float16*   sh    = ss + h * NP * NP;
  const float scale = 0.17677669529663689f;  // 32^-0.5

  // Stage 3: S = scale * Q @ K^T (raw scores, f16)
#pragma unroll
  for (int mi = 0; mi < 2; ++mi) {
    const int mt = (sub << 1) + mi;
    const int m0 = mt << 4;
    v16h a = load_a_f16(qs, m0, h * DHEAD, C3, lane);  // Q rows
    const int rbase = m0 + ((lane >> 4) << 3);
#pragma unroll
    for (int nt = 0; nt < 4; ++nt) {
      // B = K^T: (k=d, n=token) = qs[token][128 + h*32 + d]; contiguous 32B
      const int tok = (nt << 4) + (lane & 15);
      const int kb  = (lane >> 4) << 4;
      v16h bm = *(const v16h*)(qs + tok * C3 + CDIM + h * DHEAD + kb);
      v8f acc = wmma_f16(a, bm, (v8f){});
      const bool pad = tok >= WIN_N;  // key padding column (per-lane const)
#pragma unroll
      for (int r = 0; r < 8; ++r) {
        const float v = pad ? -3.0e4f : acc[r] * scale;
        sh[(rbase + r) * NP + tok] = (_Float16)v;
      }
    }
  }

  // softmax (window mask fused into first pass); lane owns row lane + 32*sub
  {
    const int row = lane + (sub << 5);
    _Float16* sr = sh + row * NP;
    if (row < WIN_N) {
      const float* mrow = mk + row * WIN_N;
      float mx = -3.0e4f;
      for (int m = 0; m < WIN_N; ++m) {
        const float v = (float)sr[m] + mrow[m];
        sr[m] = (_Float16)v;
        mx = fmaxf(mx, v);
      }
      float sum = 0.f;
      for (int m = 0; m < WIN_N; ++m) {
        const float e = __expf((float)sr[m] - mx);
        sum += e;
        sr[m] = (_Float16)e;
      }
      const float inv = 1.f / sum;
      for (int m = 0; m < WIN_N; ++m) sr[m] = (_Float16)((float)sr[m] * inv);
      for (int m = WIN_N; m < NP; ++m) sr[m] = (_Float16)0.f;  // K padding
    } else {
      for (int m = 0; m < NP; ++m) sr[m] = (_Float16)0.f;      // M padding
    }
  }

  // Stage 4: O = P @ V (K padded to 64); O -> V region of qs
  {
    v8f o[2][2];
#pragma unroll
    for (int mi = 0; mi < 2; ++mi) {
      const int mt = (sub << 1) + mi;
#pragma unroll
      for (int nt = 0; nt < 2; ++nt) o[mi][nt] = (v8f){};
#pragma unroll
      for (int k0 = 0; k0 < NP; k0 += 32) {
        v16h a = load_a_f16(sh, mt << 4, k0, NP, lane);  // P rows, f16
#pragma unroll
        for (int nt = 0; nt < 2; ++nt) {
          // B = V: (k=token, n=d) = vt[d][token]; contiguous 32B per lane
          const int d  = h * DHEAD + (nt << 4) + (lane & 15);
          const int kb = k0 + ((lane >> 4) << 4);
          v16h bm = *(const v16h*)(vt + d * NP + kb);
          o[mi][nt] = wmma_f16(a, bm, o[mi][nt]);
        }
      }
    }
#pragma unroll
    for (int mi = 0; mi < 2; ++mi) {
      const int mt    = (sub << 1) + mi;
      const int rbase = (mt << 4) + ((lane >> 4) << 3);
      const int n     = lane & 15;
#pragma unroll
      for (int nt = 0; nt < 2; ++nt)
#pragma unroll
        for (int r = 0; r < 8; ++r)
          qs[(rbase + r) * C3 + 2 * CDIM + h * DHEAD + (nt << 4) + n] =
              (_Float16)o[mi][nt][r];
    }
  }
  __syncthreads();

  // ---- Stage 5: out = O @ Wproj + bproj; wave owns 1 m-tile x 4 n-tiles ----
  {
    const _Float16* os = qs + 2 * CDIM;  // 64x128 view, ld = C3
    float* ob = out + (size_t)b * WIN_N * CDIM;
    const int mt = wave & 3;
    const int g  = wave >> 2;
    const int m0 = mt << 4;
    v16h a[4];
#pragma unroll
    for (int kt = 0; kt < 4; ++kt)
      a[kt] = load_a_f16(os, m0, kt << 5, C3, lane);
    const int rbase = m0 + ((lane >> 4) << 3);
#pragma unroll
    for (int j = 0; j < 4; ++j) {
      const int nt = (g << 2) + j;
      v8f acc = {};
#pragma unroll
      for (int kt = 0; kt < 4; ++kt)
        acc = wmma_f16(a[kt], load_b_packed(wp16, kt, nt, 4, lane), acc);
      const int   n    = (nt << 4) + (lane & 15);
      const float bias = bproj[n];
#pragma unroll
      for (int r = 0; r < 8; ++r) {
        const int row = rbase + r;
        if (row < WIN_N) ob[row * CDIM + n] = acc[r] + bias;
      }
    }
  }
}

extern "C" void kernel_launch(void* const* d_in, const int* in_sizes, int n_in,
                              void* d_out, int out_size, void* d_ws,
                              size_t ws_size, hipStream_t stream) {
  const float* x     = (const float*)d_in[0];
  const float* mask  = (const float*)d_in[1];
  const float* Wqkv  = (const float*)d_in[2];
  const float* bqkv  = (const float*)d_in[3];
  const float* Wproj = (const float*)d_in[4];
  const float* bproj = (const float*)d_in[5];
  float*       out   = (float*)d_out;

  _Float16* w16 = (_Float16*)d_ws;  // packed [128*384] Wqkv | [128*128] Wproj

  pack_weights_kernel<<<256, 256, 0, stream>>>(Wqkv, Wproj, w16);

  win_attn_kernel<<<4096, 256, LDS_BYTES, stream>>>(
      x, mask, bqkv, bproj, w16, w16 + CDIM * C3, out);
}
